// Discriminator_21947282882697
// MI455X (gfx1250) — compile-verified, matrix operand order checked by reading, not run on hardware
//
#include <hip/hip_runtime.h>
#include <hip/hip_bf16.h>
#include <math.h>

#define SEQ   4096
#define HDIM  1024
#define H3    3072
#define VOUT  50257
#define MLPH  8

typedef __attribute__((ext_vector_type(16))) __bf16 v16bf;
typedef __attribute__((ext_vector_type(4)))  __bf16 v4bf;
typedef __attribute__((ext_vector_type(8)))  float  v8f;

__device__ __forceinline__ float sigmoidf_(float x) {
    return 1.0f / (1.0f + __expf(-x));
}

// ---------------------------------------------------------------------------
// Kernel 1: embedding gather + f32 -> bf16 convert.  One block per sequence
// position; 256 threads x 4 elements = 1024 (H) per row.
// ---------------------------------------------------------------------------
__global__ __launch_bounds__(256)
void embed_bf16(const int* __restrict__ X, const float* __restrict__ emb,
                __bf16* __restrict__ xs) {
    const int s   = blockIdx.x;
    const int row = X[s];
    const int k   = threadIdx.x * 4;
    const float4 f = *reinterpret_cast<const float4*>(emb + (size_t)row * HDIM + k);
    v4bf o;
    o[0] = (__bf16)f.x; o[1] = (__bf16)f.y; o[2] = (__bf16)f.z; o[3] = (__bf16)f.w;
    *reinterpret_cast<v4bf*>(xs + (size_t)s * HDIM + k) = o;
}

// ---------------------------------------------------------------------------
// Kernel 2: bulk f32 -> bf16 convert (for W_ih).
// ---------------------------------------------------------------------------
__global__ __launch_bounds__(256)
void cvt_bf16(const float* __restrict__ src, __bf16* __restrict__ dst, int n) {
    const int i = (blockIdx.x * 256 + threadIdx.x) * 4;
    if (i < n) {
        const float4 f = *reinterpret_cast<const float4*>(src + i);
        v4bf o;
        o[0] = (__bf16)f.x; o[1] = (__bf16)f.y; o[2] = (__bf16)f.z; o[3] = (__bf16)f.w;
        *reinterpret_cast<v4bf*>(dst + i) = o;
    }
}

// ---------------------------------------------------------------------------
// Kernel 3: gi[SEQ, 3H] = xs[SEQ, H] @ W_ih^T + b_ih using
// v_wmma_f32_16x16x32_bf16.
//
// Workgroup tile 64(M) x 128(N); 8 waves arranged 2(M) x 4(N); each wave owns
// a 32x32 output patch = four 16x16 accumulators fed by 2 A-fragments and
// 2 B-fragments per K-step (8 b128 loads : 4 WMMAs, each fragment reused
// twice in registers).  Both A (xs) and "B^T" (W_ih) are row-major with K
// contiguous, so each lane fetches its two 16-byte K-chunks (ISA 16-bit A
// 16x32 striping: lanes 0-15 hold K 0-7/16-23, lanes 16-31 hold K 8-15/24-31)
// with global_load_b128.
// ---------------------------------------------------------------------------
__global__ __launch_bounds__(256)
void gemm_gi(const __bf16* __restrict__ A,      // [SEQ, H]
             const __bf16* __restrict__ B,      // [3H, H]  (W_ih, bf16)
             const float*  __restrict__ bias,   // [3H]
             float* __restrict__ C) {           // [SEQ, 3H]
    const int tid  = threadIdx.x;
    const int wave = tid >> 5;
    const int lane = tid & 31;
    const int wm   = wave & 1;               // wave row   (0..1)
    const int wn   = wave >> 1;              // wave col   (0..3)
    const int m0   = blockIdx.y * 64  + wm * 32;
    const int n0   = blockIdx.x * 128 + wn * 32;

    const int lrow  = lane & 15;
    const int khalf = lane >> 4;             // 0: K 0-7/16-23, 1: K 8-15/24-31

    v8f c00 = {}, c01 = {}, c10 = {}, c11 = {};

    const __bf16* a0Base = A + (size_t)(m0 + lrow) * HDIM;
    const __bf16* a1Base = A + (size_t)(m0 + 16 + lrow) * HDIM;
    const __bf16* b0Base = B + (size_t)(n0 + lrow) * HDIM;
    const __bf16* b1Base = B + (size_t)(n0 + 16 + lrow) * HDIM;

    for (int kb = 0; kb < HDIM; kb += 32) {
        const int o0 = kb + khalf * 8;
        const int o1 = kb + 16 + khalf * 8;
        v16bf af0, af1, bf0, bf1;
        reinterpret_cast<uint4*>(&af0)[0] = *reinterpret_cast<const uint4*>(a0Base + o0);
        reinterpret_cast<uint4*>(&af0)[1] = *reinterpret_cast<const uint4*>(a0Base + o1);
        reinterpret_cast<uint4*>(&af1)[0] = *reinterpret_cast<const uint4*>(a1Base + o0);
        reinterpret_cast<uint4*>(&af1)[1] = *reinterpret_cast<const uint4*>(a1Base + o1);
        reinterpret_cast<uint4*>(&bf0)[0] = *reinterpret_cast<const uint4*>(b0Base + o0);
        reinterpret_cast<uint4*>(&bf0)[1] = *reinterpret_cast<const uint4*>(b0Base + o1);
        reinterpret_cast<uint4*>(&bf1)[0] = *reinterpret_cast<const uint4*>(b1Base + o0);
        reinterpret_cast<uint4*>(&bf1)[1] = *reinterpret_cast<const uint4*>(b1Base + o1);

        c00 = __builtin_amdgcn_wmma_f32_16x16x32_bf16(false, af0, false, bf0,
                                                      (short)0, c00, false, false);
        c01 = __builtin_amdgcn_wmma_f32_16x16x32_bf16(false, af0, false, bf1,
                                                      (short)0, c01, false, false);
        c10 = __builtin_amdgcn_wmma_f32_16x16x32_bf16(false, af1, false, bf0,
                                                      (short)0, c10, false, false);
        c11 = __builtin_amdgcn_wmma_f32_16x16x32_bf16(false, af1, false, bf1,
                                                      (short)0, c11, false, false);
    }

    // C/D layout: lanes 0-15 -> M = r, lanes 16-31 -> M = r + 8; N = lane & 15.
    const int rowBase0 = m0 + (lane >> 4) * 8;
    const int rowBase1 = rowBase0 + 16;
    const int col0     = n0 + (lane & 15);
    const int col1     = col0 + 16;
    const float bia0   = bias[col0];
    const float bia1   = bias[col1];
#pragma unroll
    for (int r = 0; r < 8; ++r) {
        const size_t row0 = (size_t)(rowBase0 + r);
        const size_t row1 = (size_t)(rowBase1 + r);
        C[row0 * H3 + col0] = c00[r] + bia0;
        C[row0 * H3 + col1] = c01[r] + bia1;
        C[row1 * H3 + col0] = c10[r] + bia0;
        C[row1 * H3 + col1] = c11[r] + bia1;
    }
}

// ---------------------------------------------------------------------------
// Kernel 4: init h0 = 0 and the global-barrier counter.
// ---------------------------------------------------------------------------
__global__ void scan_init(float* __restrict__ hbuf, unsigned int* __restrict__ counter) {
    const int i = blockIdx.x * blockDim.x + threadIdx.x;
    if (i < HDIM) hbuf[i] = 0.0f;
    if (i == 0) *counter = 0u;
}

// ---------------------------------------------------------------------------
// Kernel 5: persistent GRU scan.  128 workgroups x 256 threads; wave w of
// block b owns hidden unit i = b*8 + w (its three W_hh rows stay hot in L2
// across all 4096 steps).  h is double-buffered in global memory, staged into
// LDS each step; cross-WGP step barrier via agent-scope atomics.
// ---------------------------------------------------------------------------
__global__ __launch_bounds__(256)
void gru_scan(const float* __restrict__ gi,    // [SEQ, 3H]
              const float* __restrict__ Whh,   // [3H, H]
              const float* __restrict__ bhh,   // [3H]
              float* __restrict__ hbuf,        // [2, H]
              unsigned int* __restrict__ counter) {
    __shared__ float hs[HDIM];
    const int tid  = threadIdx.x;
    const int wave = tid >> 5;
    const int lane = tid & 31;
    const int i    = blockIdx.x * 8 + wave;    // hidden index 0..1023

    const float* wr = Whh + (size_t)i * HDIM;
    const float* wz = Whh + (size_t)(i + HDIM) * HDIM;
    const float* wn = Whh + (size_t)(i + 2 * HDIM) * HDIM;
    const float  br = bhh[i], bz = bhh[i + HDIM], bn = bhh[i + 2 * HDIM];
    const unsigned nb = gridDim.x;

    for (int t = 0; t < SEQ; ++t) {
        float* hin  = hbuf + (size_t)(t & 1) * HDIM;
        float* hout = hbuf + (size_t)((t + 1) & 1) * HDIM;

        // Stage h_t into LDS (agent-scope loads so we never read a stale L0 line).
        for (int k = tid; k < HDIM; k += 256)
            hs[k] = __hip_atomic_load(&hin[k], __ATOMIC_RELAXED, __HIP_MEMORY_SCOPE_AGENT);
        __syncthreads();

        float pr = 0.f, pz = 0.f, pn = 0.f;
#pragma unroll 8
        for (int k = lane; k < HDIM; k += 32) {
            const float hv = hs[k];
            pr = __fmaf_rn(hv, wr[k], pr);
            pz = __fmaf_rn(hv, wz[k], pz);
            pn = __fmaf_rn(hv, wn[k], pn);
        }
#pragma unroll
        for (int m = 16; m >= 1; m >>= 1) {
            pr += __shfl_xor(pr, m, 32);
            pz += __shfl_xor(pz, m, 32);
            pn += __shfl_xor(pn, m, 32);
        }

        if (lane == 0) {
            const float* g = gi + (size_t)t * H3;
            const float r = sigmoidf_(g[i]            + pr + br);
            const float z = sigmoidf_(g[i + HDIM]     + pz + bz);
            const float n = tanhf   (g[i + 2 * HDIM] + r * (pn + bn));
            const float hnew = (1.0f - z) * n + z * hs[i];
            __hip_atomic_store(&hout[i], hnew, __ATOMIC_RELEASE, __HIP_MEMORY_SCOPE_AGENT);
            // Pull next step's gate slab toward this WGP while we sit in the
            // step barrier (global_prefetch_b8; speculative, no counters).
            __builtin_prefetch(g + H3 + i, 0, 3);
            __builtin_prefetch(g + H3 + i + HDIM, 0, 3);
            __builtin_prefetch(g + H3 + i + 2 * HDIM, 0, 3);
        }
        __syncthreads();

        // Device-wide step barrier: monotone counter, target = (t+1)*gridDim.
        if (tid == 0) {
            __hip_atomic_fetch_add(counter, 1u, __ATOMIC_RELEASE, __HIP_MEMORY_SCOPE_AGENT);
            const unsigned target = (unsigned)(t + 1) * nb;
            while (__hip_atomic_load(counter, __ATOMIC_ACQUIRE,
                                     __HIP_MEMORY_SCOPE_AGENT) < target) {
                __builtin_amdgcn_s_sleep(1);
            }
        }
        __syncthreads();
    }
}

// ---------------------------------------------------------------------------
// Kernel 6: z[8] = relu(h_last @ W1^T + b1).  One wave per gate.
// ---------------------------------------------------------------------------
__global__ __launch_bounds__(256)
void head1(const float* __restrict__ hlast, const float* __restrict__ W1,
           const float* __restrict__ b1, float* __restrict__ z8) {
    const int gate = threadIdx.x >> 5;   // 0..7
    const int lane = threadIdx.x & 31;
    const float* w = W1 + (size_t)gate * HDIM;
    float p = 0.f;
    for (int k = lane; k < HDIM; k += 32) p = __fmaf_rn(hlast[k], w[k], p);
#pragma unroll
    for (int m = 16; m >= 1; m >>= 1) p += __shfl_xor(p, m, 32);
    if (lane == 0) z8[gate] = fmaxf(p + b1[gate], 0.0f);
}

// ---------------------------------------------------------------------------
// Kernel 7: out[j] = sigmoid(z @ W2[j] + b2[j]), j < 50257.
// ---------------------------------------------------------------------------
__global__ __launch_bounds__(256)
void head2(const float* __restrict__ z8, const float* __restrict__ W2,
           const float* __restrict__ b2, float* __restrict__ out) {
    __shared__ float zs[MLPH];
    if (threadIdx.x < MLPH) zs[threadIdx.x] = z8[threadIdx.x];
    __syncthreads();
    const int j = blockIdx.x * 256 + threadIdx.x;
    if (j < VOUT) {
        const float4 w0 = *reinterpret_cast<const float4*>(W2 + (size_t)j * 8);
        const float4 w1 = *reinterpret_cast<const float4*>(W2 + (size_t)j * 8 + 4);
        float acc = b2[j];
        acc += zs[0] * w0.x + zs[1] * w0.y + zs[2] * w0.z + zs[3] * w0.w;
        acc += zs[4] * w1.x + zs[5] * w1.y + zs[6] * w1.z + zs[7] * w1.w;
        out[j] = sigmoidf_(acc);
    }
}

// ---------------------------------------------------------------------------
// Workspace layout (bytes):
//   [0,        8 MiB)  xs  bf16 [4096,1024]
//   [8 MiB,   14 MiB)  W_ih bf16 [3072,1024]
//   [16 MiB,  64 MiB)  gi  f32  [4096,3072]
//   [64 MiB, +8 KiB)   hbuf f32 [2,1024]
//   [+8 KiB, +8 KiB+32) z8 f32 [8]
//   [+8 KiB+256]       barrier counter (u32)
// ---------------------------------------------------------------------------
extern "C" void kernel_launch(void* const* d_in, const int* in_sizes, int n_in,
                              void* d_out, int out_size, void* d_ws, size_t ws_size,
                              hipStream_t stream) {
    const int*   X    = (const int*)  d_in[0];
    const float* emb  = (const float*)d_in[1];
    const float* W_ih = (const float*)d_in[2];
    const float* W_hh = (const float*)d_in[3];
    const float* b_ih = (const float*)d_in[4];
    const float* b_hh = (const float*)d_in[5];
    const float* W1   = (const float*)d_in[6];
    const float* b1   = (const float*)d_in[7];
    const float* W2   = (const float*)d_in[8];
    const float* b2   = (const float*)d_in[9];
    float* out = (float*)d_out;

    char* ws = (char*)d_ws;
    __bf16*       xs      = (__bf16*)(ws);
    __bf16*       wihb    = (__bf16*)(ws + (8u  << 20));
    float*        gi      = (float*) (ws + (16u << 20));
    float*        hbuf    = (float*) (ws + (64u << 20));
    float*        z8      = (float*) (ws + (64u << 20) + 8192);
    unsigned int* counter = (unsigned int*)(ws + (64u << 20) + 8192 + 256);

    // 1) gather + convert activations; convert W_ih to bf16
    embed_bf16<<<SEQ, 256, 0, stream>>>(X, emb, xs);
    cvt_bf16<<<(H3 * HDIM) / 1024, 256, 0, stream>>>(W_ih, wihb, H3 * HDIM);

    // 2) big parallel GEMM on the WMMA pipes (64x128 block tile, 32x32/wave)
    dim3 gemmGrid(H3 / 128, SEQ / 64);
    gemm_gi<<<gemmGrid, 256, 0, stream>>>(xs, wihb, b_ih, gi);

    // 3) persistent recurrent scan (W_hh stays L2-resident)
    scan_init<<<4, 256, 0, stream>>>(hbuf, counter);
    gru_scan<<<128, 256, 0, stream>>>(gi, W_hh, b_hh, hbuf, counter);

    // 4) head: final h lives in hbuf[0..H) (4096 is even)
    head1<<<1, 256, 0, stream>>>(hbuf, W1, b1, z8);
    head2<<<(VOUT + 255) / 256, 256, 0, stream>>>(z8, W2, b2, out);
}
